// GCN_71098888618024
// MI455X (gfx1250) — matrix operand order repaired and checked by reading, not compiled, not run on hardware
//
#include <hip/hip_runtime.h>

typedef __attribute__((ext_vector_type(16))) __bf16 v16bf;
typedef __attribute__((ext_vector_type(8)))  __bf16 v8bf;
typedef __attribute__((ext_vector_type(4)))  __bf16 v4bf;
typedef __attribute__((ext_vector_type(8)))  float  v8f;

#define N_IN_DIM  710
#define N_HID     128
#define N_OUT     64

// ---------------------------------------------------------------- init: deg=1 (self loop), gsum=0
__global__ void k_init(float* __restrict__ deg, float* __restrict__ gsum, int n) {
    int i = blockIdx.x * blockDim.x + threadIdx.x;
    if (i < n)  deg[i]  = 1.0f;
    if (i < 64) gsum[i] = 0.0f;
}

// ---------------------------------------------------------------- deg[dst] += 1 per edge
__global__ void k_degree(const int* __restrict__ dst, float* __restrict__ deg, int nE) {
    int i = blockIdx.x * blockDim.x + threadIdx.x;
    if (i < nE) atomicAdd(&deg[dst[i]], 1.0f);
}

// ---------------------------------------------------------------- dinv = rsqrt(deg)   (in place)
__global__ void k_rsqrt(float* __restrict__ d, int n) {
    int i = blockIdx.x * blockDim.x + threadIdx.x;
    if (i < n) d[i] = rsqrtf(d[i]);
}

// ---------------------------------------------------------------- W (fp32 [K][C]) -> bf16, transposed [C][Kpad], zero-padded K
template <int KDIM, int CDIM>
__global__ void k_wprep(const float* __restrict__ W, __bf16* __restrict__ WB) {
    constexpr int KP = ((KDIM + 31) / 32) * 32;
    int idx = blockIdx.x * blockDim.x + threadIdx.x;
    if (idx >= CDIM * KP) return;
    int n = idx / KP, k = idx - n * KP;
    float v = (k < KDIM) ? W[(size_t)k * CDIM + n] : 0.0f;
    WB[idx] = (__bf16)v;
}

// ---------------------------------------------------------------- bf16 WMMA GEMM (software-pipelined)
// hlin = A(fp32,[nRows x KDIM]) @ W ; WB is bf16, transposed [CDIM][Kpad].
// Also writes outInit = hlin * dinv[row]^2 (self-loop message).
// Block: 256 threads = 8 waves; tile = 128 rows x CDIM cols; K chunks of 32 staged in LDS.
template <int KDIM, int CDIM>
__global__ __launch_bounds__(256)
void k_gemm_wmma(const float* __restrict__ A, const __bf16* __restrict__ WB,
                 const float* __restrict__ dinv,
                 float* __restrict__ hlin, float* __restrict__ outInit, int nRows)
{
    constexpr int KP  = ((KDIM + 31) / 32) * 32;
    constexpr int KCH = KP / 32;
    constexpr int NT  = CDIM / 16;          // 16x16 col tiles per wave
    constexpr int NBG = CDIM / 64;          // B v8bf groups per thread per chunk (2 or 1)

    __shared__ __align__(64) __bf16 lds_a[128][32];   // [row][k]
    __shared__ __align__(64) __bf16 lds_b[CDIM][32];  // [col][k]  (already transposed)

    const int rowBase = blockIdx.x * 128;
    const bool fullRows = (rowBase + 128 <= nRows);
    const int wave = threadIdx.x >> 5;
    const int lane = threadIdx.x & 31;
    const int half = lane >> 4;
    const int l16  = lane & 15;
    const int mrow = wave * 16 + l16;       // A-fragment row (local)

    float ar[16];
    v8bf  br[NBG];

    // fetch A chunk (128x32 fp32) into regs; uniform fast path
    auto fetchA = [&](int kc) {
        const int kBase = kc * 32;
        const bool fullK = (kBase + 32 <= KDIM);
        if (fullRows && fullK) {
#pragma unroll
            for (int i = 0; i < 4; ++i) {
                int q = threadIdx.x + 256 * i;          // quad id [0,1024)
                int r = q >> 3, kq = (q & 7) * 4;
                const float* p = &A[(size_t)(rowBase + r) * KDIM + kBase + kq];
                if constexpr ((KDIM & 3) == 0) {
                    float4 v = *reinterpret_cast<const float4*>(p);
                    ar[4*i+0] = v.x; ar[4*i+1] = v.y; ar[4*i+2] = v.z; ar[4*i+3] = v.w;
                } else {                                 // rows only 8B-aligned (K=710)
                    float2 v0 = *reinterpret_cast<const float2*>(p);
                    float2 v1 = *reinterpret_cast<const float2*>(p + 2);
                    ar[4*i+0] = v0.x; ar[4*i+1] = v0.y; ar[4*i+2] = v1.x; ar[4*i+3] = v1.y;
                }
            }
        } else {
#pragma unroll
            for (int i = 0; i < 4; ++i) {
                int q = threadIdx.x + 256 * i;
                int r = q >> 3, kq = (q & 7) * 4;
                int gr = rowBase + r;
#pragma unroll
                for (int j = 0; j < 4; ++j) {
                    int gk = kBase + kq + j;
                    ar[4*i+j] = (gr < nRows && gk < KDIM) ? A[(size_t)gr * KDIM + gk] : 0.0f;
                }
            }
        }
    };
    // fetch B chunk (CDIM x 32 bf16, already transposed+padded): pure aligned copies
    auto fetchB = [&](int kc) {
#pragma unroll
        for (int i = 0; i < NBG; ++i) {
            int q = threadIdx.x + 256 * i;              // group of 8 bf16, [0, CDIM*4)
            int n = q >> 2, k8 = (q & 3) * 8;
            br[i] = *reinterpret_cast<const v8bf*>(&WB[(size_t)n * KP + kc * 32 + k8]);
        }
    };
    auto stage = [&]() {
#pragma unroll
        for (int i = 0; i < 4; ++i) {
            int q = threadIdx.x + 256 * i;
            int r = q >> 3, kq = (q & 7) * 4;
            v4bf b;
            b[0] = (__bf16)ar[4*i+0]; b[1] = (__bf16)ar[4*i+1];
            b[2] = (__bf16)ar[4*i+2]; b[3] = (__bf16)ar[4*i+3];
            *reinterpret_cast<v4bf*>(&lds_a[r][kq]) = b;
        }
#pragma unroll
        for (int i = 0; i < NBG; ++i) {
            int q = threadIdx.x + 256 * i;
            int n = q >> 2, k8 = (q & 3) * 8;
            *reinterpret_cast<v8bf*>(&lds_b[n][k8]) = br[i];
        }
    };

    v8f zero = {};
    v8f acc[NT];
#pragma unroll
    for (int c = 0; c < NT; ++c) acc[c] = zero;

    fetchA(0); fetchB(0);
    for (int kc = 0; kc < KCH; ++kc) {
        stage();
        __syncthreads();
        if (kc + 1 < KCH) { fetchA(kc + 1); fetchB(kc + 1); }  // overlap with WMMA

        // A fragment: lane holds row M=l16; K = half*16 .. +16 (contiguous, matches B)
        union { v16bf v; v8bf h[2]; } ua;
        ua.h[0] = *reinterpret_cast<const v8bf*>(&lds_a[mrow][half * 16]);
        ua.h[1] = *reinterpret_cast<const v8bf*>(&lds_a[mrow][half * 16 + 8]);
#pragma unroll
        for (int c = 0; c < NT; ++c) {
            // B fragment: lane holds col N=c*16+l16; K = half*16 .. +16 (contiguous)
            v16bf bf = *reinterpret_cast<const v16bf*>(&lds_b[c * 16 + l16][half * 16]);
            acc[c] = __builtin_amdgcn_wmma_f32_16x16x32_bf16(
                false, ua.v, false, bf, (short)0, acc[c], false, false);
        }
        __syncthreads();
    }

    // Epilogue: D layout -> row M = j + 8*half, col N = l16
    const int gr0 = rowBase + wave * 16 + 8 * half;
    if (fullRows) {
        float dv2[8];
#pragma unroll
        for (int j = 0; j < 8; ++j) { float di = dinv[gr0 + j]; dv2[j] = di * di; }
#pragma unroll
        for (int c = 0; c < NT; ++c) {
#pragma unroll
            for (int j = 0; j < 8; ++j) {
                float v = acc[c][j];
                size_t idx = (size_t)(gr0 + j) * CDIM + c * 16 + l16;
                hlin[idx]    = v;
                outInit[idx] = v * dv2[j];
            }
        }
    } else {
        float dv2[8];
#pragma unroll
        for (int j = 0; j < 8; ++j) {
            int r = gr0 + j;
            float di = (r < nRows) ? dinv[r] : 0.0f;
            dv2[j] = di * di;
        }
#pragma unroll
        for (int c = 0; c < NT; ++c) {
#pragma unroll
            for (int j = 0; j < 8; ++j) {
                int r = gr0 + j;
                if (r < nRows) {
                    float v = acc[c][j];
                    size_t idx = (size_t)r * CDIM + c * 16 + l16;
                    hlin[idx]    = v;
                    outInit[idx] = v * dv2[j];
                }
            }
        }
    }
}

// ---------------------------------------------------------------- edge scatter: out[dst] += h[src]*dinv[s]*dinv[d]
template <int CDIM>
__global__ __launch_bounds__(256)
void k_edge_scatter(const int* __restrict__ src, const int* __restrict__ dst,
                    const float* __restrict__ dinv, const float* __restrict__ h,
                    float* __restrict__ out, int nE)
{
    const int lane = threadIdx.x & 31;
    int wid = (blockIdx.x * blockDim.x + threadIdx.x) >> 5;
    const int nW = (gridDim.x * blockDim.x) >> 5;
    for (int e = wid; e < nE; e += nW) {
        const int s = src[e], d = dst[e];
        const float nrm = dinv[s] * dinv[d];
        if constexpr (CDIM == 128) {
            const float4 v = *reinterpret_cast<const float4*>(&h[(size_t)s * 128 + lane * 4]);
            float* op = &out[(size_t)d * 128 + lane * 4];
            atomicAdd(op + 0, v.x * nrm);
            atomicAdd(op + 1, v.y * nrm);
            atomicAdd(op + 2, v.z * nrm);
            atomicAdd(op + 3, v.w * nrm);
        } else {
            const float2 v = *reinterpret_cast<const float2*>(&h[(size_t)s * 64 + lane * 2]);
            float* op = &out[(size_t)d * 64 + lane * 2];
            atomicAdd(op + 0, v.x * nrm);
            atomicAdd(op + 1, v.y * nrm);
        }
    }
}

// ---------------------------------------------------------------- h = relu(h + b); optional column-sum for mean
template <int CDIM, bool MEAN>
__global__ __launch_bounds__(256)
void k_bias_relu(float* __restrict__ h, const float* __restrict__ b,
                 float* __restrict__ gsum, size_t total)
{
    __shared__ float ls[CDIM];
    if (MEAN) {
        if (threadIdx.x < CDIM) ls[threadIdx.x] = 0.0f;
        __syncthreads();
    }
    const size_t stride = (size_t)gridDim.x * blockDim.x;   // multiple of CDIM
    size_t idx = (size_t)blockIdx.x * blockDim.x + threadIdx.x;
    float local = 0.0f;
    const int mycol = (int)(idx & (CDIM - 1));              // invariant across iterations
    for (; idx < total; idx += stride) {
        float v = h[idx] + b[mycol];
        v = fmaxf(v, 0.0f);
        h[idx] = v;
        if (MEAN) local += v;
    }
    if (MEAN) {
        atomicAdd(&ls[mycol], local);
        __syncthreads();
        if (threadIdx.x < CDIM) atomicAdd(&gsum[threadIdx.x], ls[threadIdx.x]);
    }
}

// ---------------------------------------------------------------- out[0] = (gsum/N) . fcW + fcb
__global__ void k_final_fc(const float* __restrict__ gsum, const float* __restrict__ fcW,
                           const float* __restrict__ fcb, float* __restrict__ out, float invN)
{
    __shared__ float ls[64];
    const int t = threadIdx.x;
    ls[t] = gsum[t] * invN * fcW[t];
    __syncthreads();
    if (t == 0) {
        float s = 0.0f;
#pragma unroll
        for (int i = 0; i < 64; ++i) s += ls[i];
        out[0] = s + fcb[0];
    }
}

// ----------------------------------------------------------------
extern "C" void kernel_launch(void* const* d_in, const int* in_sizes, int n_in,
                              void* d_out, int out_size, void* d_ws, size_t ws_size,
                              hipStream_t stream)
{
    const float* x    = (const float*)d_in[0];
    const int*   eidx = (const int*)  d_in[1];
    const float* W1   = (const float*)d_in[2];
    const float* b1   = (const float*)d_in[3];
    const float* W2   = (const float*)d_in[4];
    const float* b2   = (const float*)d_in[5];
    const float* fcW  = (const float*)d_in[6];
    const float* fcb  = (const float*)d_in[7];
    float* out = (float*)d_out;

    const int N = in_sizes[0] / N_IN_DIM;       // 100000
    const int E = in_sizes[1] / 2;              // 1600000
    const int* src = eidx;
    const int* dst = eidx + E;

    // workspace layout (floats); h2lin/out2 alias the dead h1lin region
    const size_t Npad = ((size_t)N + 255) & ~(size_t)255;
    float* ws    = (float*)d_ws;
    float* deg   = ws;                                   // N   (becomes dinv)
    float* gsum  = ws + Npad;                            // 64
    float* h1lin = ws + Npad + 256;                      // N*128
    float* out1  = h1lin + (size_t)N * 128;              // N*128
    float* h2lin = h1lin;                                // N*64  (aliases dead h1lin)
    float* out2  = h1lin + (size_t)N * 64;               // N*64
    constexpr int KP1 = ((N_IN_DIM + 31) / 32) * 32;     // 736
    constexpr int KP2 = N_HID;                           // 128
    __bf16* wbf1 = (__bf16*)(out1 + (size_t)N * 128);    // [128][736] bf16
    __bf16* wbf2 = wbf1 + (size_t)N_HID * KP1;           // [64][128]  bf16

    const int T = 256;
    const int gN  = (N + T - 1) / T;
    const int gE  = (E + T - 1) / T;
    const int gMM = (N + 127) / 128;

    // weight preconversion (bf16, transposed, K-padded) + degrees/norms
    k_wprep<N_IN_DIM, N_HID><<<(N_HID * KP1 + T - 1) / T, T, 0, stream>>>(W1, wbf1);
    k_wprep<N_HID,    N_OUT><<<(N_OUT * KP2 + T - 1) / T, T, 0, stream>>>(W2, wbf2);
    k_init  <<<gN, T, 0, stream>>>(deg, gsum, N);
    k_degree<<<gE, T, 0, stream>>>(dst, deg, E);
    k_rsqrt <<<gN, T, 0, stream>>>(deg, N);

    // layer 1: GEMM (WMMA bf16) -> scatter -> bias+relu
    k_gemm_wmma<N_IN_DIM, N_HID><<<gMM, T, 0, stream>>>(x, wbf1, deg, h1lin, out1, N);
    k_edge_scatter<N_HID><<<4096, T, 0, stream>>>(src, dst, deg, h1lin, out1, E);
    k_bias_relu<N_HID, false><<<4096, T, 0, stream>>>(out1, b1, nullptr, (size_t)N * N_HID);

    // layer 2: GEMM (WMMA bf16) -> scatter -> bias+relu + column sums
    k_gemm_wmma<N_HID, N_OUT><<<gMM, T, 0, stream>>>(out1, wbf2, deg, h2lin, out2, N);
    k_edge_scatter<N_OUT><<<4096, T, 0, stream>>>(src, dst, deg, h2lin, out2, E);
    k_bias_relu<N_OUT, true><<<4096, T, 0, stream>>>(out2, b2, gsum, (size_t)N * N_OUT);

    // readout + fc
    k_final_fc<<<1, 64, 0, stream>>>(gsum, fcW, fcb, out, 1.0f / (float)N);
}
